// TransformerCrossBlock_6751688589659
// MI455X (gfx1250) — matrix operand order repaired and checked by reading, not compiled, not run on hardware
//
#include <hip/hip_runtime.h>

// ---------------- problem constants ----------------
#define DIMC   512
#define HEADS  8
#define DEPTHC 64
#define DFFC   2048
#define SEQ    384
#define BATCH  32
#define MROWS  (BATCH * SEQ)          // 12288
#define MD     ((size_t)MROWS * DIMC) // 6,291,456 elements

#define LDG 40   // GEMM LDS row stride (f16): 80B, 16B aligned, conflict-friendly
#define LDA 88   // attention LDS row stride (f16): 176B, 16B aligned, conflict-free

typedef __attribute__((ext_vector_type(16))) _Float16 v16h;
typedef __attribute__((ext_vector_type(8)))  float    v8f;
typedef unsigned int u32x4 __attribute__((ext_vector_type(4)));
typedef int          i32x8 __attribute__((ext_vector_type(8)));
typedef int          i32x4 __attribute__((ext_vector_type(4)));

// ---------------- WMMA fragment loaders (gfx1250 wave32 layouts) ----------------
// Contiguous loader: works for
//  * B-frag from [n][k] storage (this code's k order: 16*half + s per lane), and
//  * A-frag from natural [m][k] storage (same k order as B).
// Each lane reads 32 contiguous bytes = two b128 LDS loads.
static __device__ __forceinline__ v16h frag_ldsc(const _Float16* base, int ld) {
  const int lane = threadIdx.x & 31;
  const _Float16* p = base + (lane & 15) * ld + (lane >> 4) * 16;
  union { v16h v; uint4 q[2]; } r;
  r.q[0] = *(const uint4*)p;
  r.q[1] = *(const uint4*)(p + 8);
  return r.v;
}

// Strided A-frag straight from global (row-major, un-permuted); used once per
// block for Q in attention.  dword j holds K = 2j + 16*half.
static __device__ __forceinline__ v16h frag_a_g(const _Float16* base, int ld) {
  const int lane = threadIdx.x & 31;
  const int hl = lane >> 4, m = lane & 15;
  const _Float16* row = base + m * ld;
  union { v16h v; unsigned u[8]; } r;
#pragma unroll
  for (int j = 0; j < 8; ++j) {
    const int k = 2 * j + 16 * hl;
    r.u[j] = *(const unsigned*)(row + k);
  }
  return r.v;
}

// ---------------- TDM helper: DMA a 2D f16 tile [rows x 32] into LDS ----------
// One descriptor per wave; LDS rows padded 16dw + 4dw -> stride LDG (40 f16).
// D# group0: count=1 | lds_addr | global_addr | type=2
// D# group1: data_size=1(2B), pad_enable, pad_interval=3(16dw), pad_amount=3(4dw),
//            tensor_dim0=32, tensor_dim1=rows, tile_dim0=32, tile_dim1=rows,
//            tensor_dim0_stride=K elements.
static __device__ __forceinline__ void tdm_load_tile32(const _Float16* gsrc,
                                                       void* lds_dst,
                                                       int rows, int K) {
  unsigned long long ga = (unsigned long long)(const void*)gsrc;
  unsigned lo = (unsigned)(size_t)lds_dst;      // generic LDS ptr: low 32 bits = LDS offset
  u32x4 g0;
  g0.x = 1u;                                    // count = 1 valid descriptor
  g0.y = lo;                                    // lds_addr
  g0.z = (unsigned)ga;                          // global_addr[31:0]
  g0.w = (unsigned)((ga >> 32) & 0x01FFFFFFu) | 0x80000000u;  // addr[56:32] | type=2
  i32x8 g1;
  g1[0] = (1 << 16) | (1 << 20) | (3 << 22) | (3 << 25);      // dsize|pad_en|int|amt
  g1[1] = 32 << 16;                             // tensor_dim0 = 32   (bits 79:48)
  g1[2] = rows << 16;                           // tensor_dim1 = rows (bits 111:80)
  g1[3] = 32 << 16;                             // tile_dim0 = 32     (bits 127:112)
  g1[4] = rows;                                 // tile_dim1 = rows   (bits 143:128)
  g1[5] = K;                                    // tensor_dim0_stride (bits 207:160)
  g1[6] = 0;
  g1[7] = 0;
  i32x4 z4 = {0, 0, 0, 0};
  i32x8 z8 = {0, 0, 0, 0, 0, 0, 0, 0};
  // 6-arg toolchain variant: (g0, g1, g2, g3, g4, cpol)
  __builtin_amdgcn_tensor_load_to_lds(g0, g1, z4, z4, z8, 0);
}

// ---------------- generic tiled WMMA GEMM with fused epilogues ----------------
// C[M,N] = A[M,K] (f16) @ W^T (f16, stored transposed [N][K]), f32 accumulate.
// MODE 0: +bias        -> f16 store      (QKV projections)
// MODE 2: +bias +resid -> f32 store      (O projections + residual)
// MODE 3: relu         -> f16 store      (FF1)
// MODE 4: +resid, BN4  -> f32 store      (FF2 + residual + batchnorm -> d_out)
// 4 waves / block; each wave owns a 64x64 tile = 16 v_wmma per 32-wide K-step.
// B tile is DMA'd by the Tensor Data Mover (one 32-row descriptor per wave);
// A tile is staged by 16B vector copies in natural k order.
template<int MODE>
__global__ __launch_bounds__(128) void gemm_wmma(
    const _Float16* __restrict__ A, const _Float16* __restrict__ Wt,
    const float* __restrict__ bias, const float* __restrict__ resid,
    const float* __restrict__ bg, const float* __restrict__ bb,
    const float* __restrict__ bm, const float* __restrict__ bv,
    float* __restrict__ outF, _Float16* __restrict__ outH,
    int M, int N, int K)
{
  __shared__ _Float16 lA[128 * LDG];   // [row][k] (natural)
  __shared__ _Float16 lB[128 * LDG];   // [n][k] (natural, weights pre-transposed)

  const int tid = threadIdx.x;
  const int wave = tid >> 5, lane = tid & 31, hl = lane >> 4, ln = lane & 15;
  const int wm = wave >> 1, wn = wave & 1;                 // 2x2 wave grid
  const int m0 = blockIdx.y * 128, n0 = blockIdx.x * 128;

  v8f acc[4][4];
#pragma unroll
  for (int i = 0; i < 4; ++i)
#pragma unroll
    for (int j = 0; j < 4; ++j)
#pragma unroll
      for (int r = 0; r < 8; ++r) acc[i][j][r] = 0.f;

  const int segPerm[4] = {0, 8, 16, 24};  // A K-chunk placement (octs of 8 f16, natural order)

  for (int k0 = 0; k0 < K; k0 += 32) {
    const bool more = (k0 + 32) < K;
    // --- B tile via TDM: each wave DMAs 32 rows of W^T (wave-uniform descriptor)
    tdm_load_tile32(Wt + (size_t)(n0 + wave * 32) * K + k0,
                    (void*)&lB[wave * 32 * LDG], 32, K);
    // --- A tile (128x32, natural k order) as 16B vectors
#pragma unroll
    for (int it = 0; it < 4; ++it) {
      int idx = tid + it * 128;               // 512 octs
      int row = idx >> 2, seg = idx & 3;
      const _Float16* pA = A + (size_t)(m0 + row) * K + k0 + seg * 8;
      *(uint4*)&lA[row * LDG + segPerm[seg]] = *(const uint4*)pA;
      if (more) __builtin_prefetch(pA + 32, 0, 1);
    }
    __builtin_amdgcn_s_wait_tensorcnt(0);     // own wave's DMA complete
    __syncthreads();                          // publish lA + all waves' lB slices

    v16h af[4];
#pragma unroll
    for (int mt = 0; mt < 4; ++mt) af[mt] = frag_ldsc(lA + (wm * 64 + mt * 16) * LDG, LDG);
#pragma unroll
    for (int nt = 0; nt < 4; ++nt) {
      v16h bf = frag_ldsc(lB + (wn * 64 + nt * 16) * LDG, LDG);
#pragma unroll
      for (int mt = 0; mt < 4; ++mt)
        acc[mt][nt] = __builtin_amdgcn_wmma_f32_16x16x32_f16(
            false, af[mt], false, bf, (short)0, acc[mt][nt], false, false);
    }
    __syncthreads();
  }

  // epilogue (C layout: VGPR r -> row r+8*hl, lane -> col ln)
#pragma unroll
  for (int mt = 0; mt < 4; ++mt)
#pragma unroll
    for (int nt = 0; nt < 4; ++nt)
#pragma unroll
      for (int r = 0; r < 8; ++r) {
        int row = m0 + wm * 64 + mt * 16 + r + 8 * hl;
        int col = n0 + wn * 64 + nt * 16 + ln;
        size_t o = (size_t)row * N + col;
        float v = acc[mt][nt][r];
        if (MODE == 0)      { v += bias[col];              outH[o] = (_Float16)v; }
        else if (MODE == 2) { v += bias[col] + resid[o];   outF[o] = v; }
        else if (MODE == 3) { v = fmaxf(v, 0.f);           outH[o] = (_Float16)v; }
        else {              // MODE 4: residual + BN4
          v += resid[o];
          float s = bg[col] * rsqrtf(bv[col] + 1e-3f);
          outF[o] = (v - bm[col]) * s + bb[col];
        }
      }
}

// ---------------- WMMA flash attention ----------------
// block = (qblk of 128 rows, head, batch); 8 waves x 16 query rows each.
template<bool CAUSAL>
__global__ __launch_bounds__(256) void flash_attn(
    const _Float16* __restrict__ Qm, const _Float16* __restrict__ Km,
    const _Float16* __restrict__ Vm, _Float16* __restrict__ Om, int S)
{
  __shared__ _Float16 lK [64 * LDA];       // [key][depth]   (B-frag for Q@K^T)
  __shared__ _Float16 lVt[64 * LDA];       // [depth][key]   (B-frag for P@V)
  __shared__ _Float16 lP [8 * 16 * LDA];   // per-wave P strip [row][key]

  const int b = blockIdx.z, h = blockIdx.y, qblk = blockIdx.x;
  const int tid = threadIdx.x, wave = tid >> 5, lane = tid & 31;
  const int hl = lane >> 4, ln = lane & 15;
  const int qrow0 = qblk * 128 + wave * 16;
  _Float16* lPw = lP + wave * 16 * LDA;

  // Q fragments: 16 rows x 64 depth -> 2 A-frags, read straight from global
  v16h qf[2];
#pragma unroll
  for (int kc = 0; kc < 2; ++kc)
    qf[kc] = frag_a_g(Qm + (size_t)(b * S + qrow0) * DIMC + h * DEPTHC + kc * 32, DIMC);

  v8f oacc[4];
  float mrow[8], lrow[8];
#pragma unroll
  for (int dt = 0; dt < 4; ++dt)
#pragma unroll
    for (int r = 0; r < 8; ++r) oacc[dt][r] = 0.f;
#pragma unroll
  for (int r = 0; r < 8; ++r) { mrow[r] = -1e30f; lrow[r] = 0.f; }

  const int nkb = CAUSAL ? 2 * (qblk + 1) : S / 64;
  for (int kb = 0; kb < nkb; ++kb) {
    const int key0 = kb * 64;
    // stage K block [key][depth] (vectorized) and V transposed [depth][key]
#pragma unroll
    for (int it = 0; it < 2; ++it) {
      int idx = tid + it * 256;              // 512 octs
      int row = idx >> 3, seg = idx & 7;
      uint4 u = *(const uint4*)(Km + (size_t)(b * S + key0 + row) * DIMC + h * DEPTHC + seg * 8);
      *(uint4*)&lK[row * LDA + seg * 8] = u;
    }
#pragma unroll
    for (int it = 0; it < 2; ++it) {
      int idx = tid + it * 256;
      int key = idx >> 3, seg = idx & 7;
      union { uint4 u; _Float16 hh[8]; } c;
      c.u = *(const uint4*)(Vm + (size_t)(b * S + key0 + key) * DIMC + h * DEPTHC + seg * 8);
#pragma unroll
      for (int q = 0; q < 8; ++q) lVt[(seg * 8 + q) * LDA + key] = c.hh[q];
    }
    __syncthreads();

    // logits: 16 q-rows x 64 keys = 4 WMMA tiles, K-dim = depth (2 chunks)
    v8f st[4];
#pragma unroll
    for (int nt = 0; nt < 4; ++nt) {
      v8f z;
#pragma unroll
      for (int r = 0; r < 8; ++r) z[r] = 0.f;
      z = __builtin_amdgcn_wmma_f32_16x16x32_f16(false, qf[0], false,
            frag_ldsc(lK + nt * 16 * LDA, LDA), (short)0, z, false, false);
      z = __builtin_amdgcn_wmma_f32_16x16x32_f16(false, qf[1], false,
            frag_ldsc(lK + nt * 16 * LDA + 32, LDA), (short)0, z, false, false);
      st[nt] = z;
    }

    // online softmax update (row stats via 16-lane shfl_xor reductions)
#pragma unroll
    for (int r = 0; r < 8; ++r) {
      const int srow = qrow0 + r + 8 * hl;
      float rowv = -1e30f;
#pragma unroll
      for (int nt = 0; nt < 4; ++nt) {
        float sv = st[nt][r] * 0.125f;                       // 1/sqrt(64)
        if (CAUSAL && (key0 + nt * 16 + ln) > srow) sv = -1e30f;
        st[nt][r] = sv;
        rowv = fmaxf(rowv, sv);
      }
#pragma unroll
      for (int msk = 1; msk < 16; msk <<= 1) rowv = fmaxf(rowv, __shfl_xor(rowv, msk));
      const float mnew = fmaxf(mrow[r], rowv);
      const float alpha = __expf(mrow[r] - mnew);
      mrow[r] = mnew;
      float rs = 0.f;
#pragma unroll
      for (int nt = 0; nt < 4; ++nt) {
        float p = __expf(st[nt][r] - mnew);
        rs += p;
        // store in natural key order so the A-frag reload is contiguous b128s
        int c = nt * 16 + ln;
        int cm = c & 24;
        int pc = c;
        lPw[(r + 8 * hl) * LDA + pc] = (_Float16)p;
      }
#pragma unroll
      for (int msk = 1; msk < 16; msk <<= 1) rs += __shfl_xor(rs, msk);
      lrow[r] = lrow[r] * alpha + rs;
#pragma unroll
      for (int dt = 0; dt < 4; ++dt) oacc[dt][r] *= alpha;
    }

    // P @ V : K-dim = 64 keys (2 chunks), 4 depth tiles
#pragma unroll
    for (int kc = 0; kc < 2; ++kc) {
      v16h pa = frag_ldsc(lPw + kc * 32, LDA);
#pragma unroll
      for (int dt = 0; dt < 4; ++dt)
        oacc[dt] = __builtin_amdgcn_wmma_f32_16x16x32_f16(false, pa, false,
                     frag_ldsc(lVt + dt * 16 * LDA + kc * 32, LDA), (short)0, oacc[dt], false, false);
    }
    __syncthreads();
  }

  // normalize + store f16 head slice
#pragma unroll
  for (int r = 0; r < 8; ++r) {
    float inv = 1.f / lrow[r];
    int row = qrow0 + r + 8 * hl;
#pragma unroll
    for (int dt = 0; dt < 4; ++dt)
      Om[(size_t)(b * S + row) * DIMC + h * DEPTHC + dt * 16 + ln] =
          (_Float16)(oacc[dt][r] * inv);
  }
}

// ---------------- elementwise helpers ----------------
__global__ __launch_bounds__(256) void cvt_f16(const float* __restrict__ in,
                                               _Float16* __restrict__ out, int n) {
  int i = (blockIdx.x * 256 + threadIdx.x) * 4;
  if (i < n) {
    float4 v = *(const float4*)(in + i);
    out[i]     = (_Float16)v.x;  out[i + 1] = (_Float16)v.y;
    out[i + 2] = (_Float16)v.z;  out[i + 3] = (_Float16)v.w;
  }
}

// transpose-convert: in[K][N] f32 -> out[N][K] f16 (LDS-tiled, coalesced both sides)
__global__ __launch_bounds__(256) void cvt_f16_T(const float* __restrict__ in,
                                                 _Float16* __restrict__ out, int K, int N) {
  __shared__ float t[32][33];
  const int tx = threadIdx.x & 31, ty = threadIdx.x >> 5;   // 32 x 8
  const int bn = blockIdx.x * 32, bk = blockIdx.y * 32;
#pragma unroll
  for (int i = 0; i < 4; ++i)
    t[ty + i * 8][tx] = in[(size_t)(bk + ty + i * 8) * N + bn + tx];
  __syncthreads();
#pragma unroll
  for (int i = 0; i < 4; ++i)
    out[(size_t)(bn + ty + i * 8) * K + bk + tx] = (_Float16)t[tx][ty + i * 8];
}

// out = bn(x) as f16 ; C = 512 (power of two)
__global__ __launch_bounds__(256) void bn_f16(
    const float* __restrict__ x, const float* __restrict__ g, const float* __restrict__ be,
    const float* __restrict__ mu, const float* __restrict__ va,
    _Float16* __restrict__ out, int n) {
  int i = (blockIdx.x * 256 + threadIdx.x) * 4;
  if (i < n) {
    float4 v = *(const float4*)(x + i);
    float vv[4] = {v.x, v.y, v.z, v.w};
#pragma unroll
    for (int q = 0; q < 4; ++q) {
      int c = (i + q) & (DIMC - 1);
      float s = g[c] * rsqrtf(va[c] + 1e-3f);
      out[i + q] = (_Float16)((vv[q] - mu[c]) * s + be[c]);
    }
  }
}

// ---------------- host orchestration ----------------
extern "C" void kernel_launch(void* const* d_in, const int* in_sizes, int n_in,
                              void* d_out, int out_size, void* d_ws, size_t ws_size,
                              hipStream_t stream) {
  const float* enc = (const float*)d_in[0];
  const float* x2  = (const float*)d_in[1];
  const float *Wq1 = (const float*)d_in[2],  *bq1 = (const float*)d_in[3];
  const float *Wk1 = (const float*)d_in[4],  *bk1 = (const float*)d_in[5];
  const float *Wv1 = (const float*)d_in[6],  *bv1 = (const float*)d_in[7];
  const float *Wo1 = (const float*)d_in[8],  *bo1 = (const float*)d_in[9];
  const float *Wq2 = (const float*)d_in[10], *bq2 = (const float*)d_in[11];
  const float *Wk2 = (const float*)d_in[12], *bk2 = (const float*)d_in[13];
  const float *Wv2 = (const float*)d_in[14], *bv2 = (const float*)d_in[15];
  const float *Wo2 = (const float*)d_in[16], *bo2 = (const float*)d_in[17];
  const float *Wff1 = (const float*)d_in[18], *Wff2 = (const float*)d_in[19];
  const float *b1g = (const float*)d_in[20], *b1b = (const float*)d_in[21],
              *b1m = (const float*)d_in[22], *b1v = (const float*)d_in[23];
  const float *b2g = (const float*)d_in[24], *b2b = (const float*)d_in[25],
              *b2m = (const float*)d_in[26], *b2v = (const float*)d_in[27];
  const float *b3g = (const float*)d_in[28], *b3b = (const float*)d_in[29],
              *b3m = (const float*)d_in[30], *b3v = (const float*)d_in[31];
  const float *b4g = (const float*)d_in[32], *b4b = (const float*)d_in[33],
              *b4m = (const float*)d_in[34], *b4v = (const float*)d_in[35];
  float* out = (float*)d_out;

  // workspace carve-up
  char* w = (char*)d_ws;
  auto take = [&](size_t bytes) { void* p = (void*)w; w += (bytes + 255) & ~(size_t)255; return p; };
  _Float16* xb  = (_Float16*)take(MD * 2);                   // f16(x2)
  _Float16* qb  = (_Float16*)take(MD * 2);
  _Float16* kb  = (_Float16*)take(MD * 2);
  _Float16* vb  = (_Float16*)take(MD * 2);
  _Float16* ab  = (_Float16*)take(MD * 2);                   // attn output (concat heads)
  _Float16* xnb = (_Float16*)take(MD * 2);                   // bn1/bn3 output
  _Float16* yb  = (_Float16*)take(MD * 2);                   // bn2(enc)
  float*    xf  = (float*)take(MD * 4);                      // x / cross_out (f32 residual carrier)
  _Float16* hb  = (_Float16*)take((size_t)MROWS * DFFC * 2); // FFN hidden
  _Float16* wq1h = (_Float16*)take(DIMC * DIMC * 2);         // all weights stored TRANSPOSED [N][K]
  _Float16* wk1h = (_Float16*)take(DIMC * DIMC * 2);
  _Float16* wv1h = (_Float16*)take(DIMC * DIMC * 2);
  _Float16* wo1h = (_Float16*)take(DIMC * DIMC * 2);
  _Float16* wq2h = (_Float16*)take(DIMC * DIMC * 2);
  _Float16* wk2h = (_Float16*)take(DIMC * DIMC * 2);
  _Float16* wv2h = (_Float16*)take(DIMC * DIMC * 2);
  _Float16* wo2h = (_Float16*)take(DIMC * DIMC * 2);
  _Float16* wf1h = (_Float16*)take((size_t)DIMC * DFFC * 2); // [2048][512]
  _Float16* wf2h = (_Float16*)take((size_t)DFFC * DIMC * 2); // [512][2048]

  const int nMD = (int)MD;                      // 6291456
  const dim3 tT512(DIMC / 32, DIMC / 32);       // (16,16)
  const dim3 tTff1(DFFC / 32, DIMC / 32);       // (64,16)  in[512][2048]
  const dim3 tTff2(DIMC / 32, DFFC / 32);       // (16,64)  in[2048][512]

  // f32 -> f16 transposed weight conversions
  cvt_f16_T<<<tT512, 256, 0, stream>>>(Wq1, wq1h, DIMC, DIMC);
  cvt_f16_T<<<tT512, 256, 0, stream>>>(Wk1, wk1h, DIMC, DIMC);
  cvt_f16_T<<<tT512, 256, 0, stream>>>(Wv1, wv1h, DIMC, DIMC);
  cvt_f16_T<<<tT512, 256, 0, stream>>>(Wo1, wo1h, DIMC, DIMC);
  cvt_f16_T<<<tT512, 256, 0, stream>>>(Wq2, wq2h, DIMC, DIMC);
  cvt_f16_T<<<tT512, 256, 0, stream>>>(Wk2, wk2h, DIMC, DIMC);
  cvt_f16_T<<<tT512, 256, 0, stream>>>(Wv2, wv2h, DIMC, DIMC);
  cvt_f16_T<<<tT512, 256, 0, stream>>>(Wo2, wo2h, DIMC, DIMC);
  cvt_f16_T<<<tTff1, 256, 0, stream>>>(Wff1, wf1h, DIMC, DFFC);
  cvt_f16_T<<<tTff2, 256, 0, stream>>>(Wff2, wf2h, DFFC, DIMC);
  cvt_f16<<<nMD / 1024, 256, 0, stream>>>(x2, xb, nMD);

  const dim3 g512(DIMC / 128, MROWS / 128);     // (4, 96)
  const dim3 gFF(DFFC / 128, MROWS / 128);      // (16, 96)
  const dim3 gAttn(SEQ / 128, HEADS, BATCH);    // (3, 8, 32)

  // ---- self attention (causal) ----
  gemm_wmma<0><<<g512, 128, 0, stream>>>(xb, wq1h, bq1, nullptr, nullptr, nullptr, nullptr, nullptr,
                                         nullptr, qb, MROWS, DIMC, DIMC);
  gemm_wmma<0><<<g512, 128, 0, stream>>>(xb, wk1h, bk1, nullptr, nullptr, nullptr, nullptr, nullptr,
                                         nullptr, kb, MROWS, DIMC, DIMC);
  gemm_wmma<0><<<g512, 128, 0, stream>>>(xb, wv1h, bv1, nullptr, nullptr, nullptr, nullptr, nullptr,
                                         nullptr, vb, MROWS, DIMC, DIMC);
  flash_attn<true><<<gAttn, 256, 0, stream>>>(qb, kb, vb, ab, SEQ);
  gemm_wmma<2><<<g512, 128, 0, stream>>>(ab, wo1h, bo1, x2, nullptr, nullptr, nullptr, nullptr,
                                         xf, nullptr, MROWS, DIMC, DIMC);   // x = o1 + x2

  // ---- batchnorms feeding cross attention ----
  bn_f16<<<nMD / 1024, 256, 0, stream>>>(xf, b1g, b1b, b1m, b1v, xnb, nMD); // xn = bn1(x)
  bn_f16<<<nMD / 1024, 256, 0, stream>>>(enc, b2g, b2b, b2m, b2v, yb, nMD); // y  = bn2(enc)

  // ---- cross attention ----
  gemm_wmma<0><<<g512, 128, 0, stream>>>(xnb, wq2h, bq2, nullptr, nullptr, nullptr, nullptr, nullptr,
                                         nullptr, qb, MROWS, DIMC, DIMC);
  gemm_wmma<0><<<g512, 128, 0, stream>>>(yb, wk2h, bk2, nullptr, nullptr, nullptr, nullptr, nullptr,
                                         nullptr, kb, MROWS, DIMC, DIMC);
  gemm_wmma<0><<<g512, 128, 0, stream>>>(yb, wv2h, bv2, nullptr, nullptr, nullptr, nullptr, nullptr,
                                         nullptr, vb, MROWS, DIMC, DIMC);
  flash_attn<false><<<gAttn, 256, 0, stream>>>(qb, kb, vb, ab, SEQ);
  gemm_wmma<2><<<g512, 128, 0, stream>>>(ab, wo2h, bo2, x2, nullptr, nullptr, nullptr, nullptr,
                                         xf, nullptr, MROWS, DIMC, DIMC);   // cross = o2 + x2

  // ---- FFN + final BN (fused) ----
  bn_f16<<<nMD / 1024, 256, 0, stream>>>(xf, b3g, b3b, b3m, b3v, xnb, nMD); // bn3(cross)
  gemm_wmma<3><<<gFF, 128, 0, stream>>>(xnb, wf1h, nullptr, nullptr, nullptr, nullptr, nullptr, nullptr,
                                        nullptr, hb, MROWS, DFFC, DIMC);    // h = relu(.@Wff1)
  gemm_wmma<4><<<g512, 128, 0, stream>>>(hb, wf2h, nullptr, xf, b4g, b4b, b4m, b4v,
                                         out, nullptr, MROWS, DIMC, DFFC);  // bn4(h@Wff2 + cross)
}
